// PartitionAttentionCl_644245094963
// MI455X (gfx1250) — matrix operand order repaired and checked by reading, not compiled, not run on hardware
//
#include <hip/hip_runtime.h>
#include <hip/hip_bf16.h>
#include <math.h>

// ---------------------------------------------------------------------------
// Swin-v2 style PartitionAttention block for MI455X (gfx1250, wave32, WMMA).
// All GEMMs: f16 inputs, f32 accumulation via v_wmma_f32_16x16x32_f16.
// Weights converted once to f16 (stay resident in 192MB L2); activations
// streamed with per-wave 32x64 output tiles + register double-buffering so
// loads for step k+1 overlap the 8 WMMAs of step k. Attention fused per
// (window,head) in LDS. Epilogues fuse bias / residual+layerscale / GLU gate.
// ---------------------------------------------------------------------------

typedef __attribute__((ext_vector_type(16))) _Float16 v16h;
typedef __attribute__((ext_vector_type(8)))  _Float16 v8h;
typedef __attribute__((ext_vector_type(8)))  float    v8f;

#define DIMC   384
#define NHEADS 12
#define HDIM   32
#define WHGT   8
#define WWID   10
#define NTOK   80                 // tokens per window (= 5 * 16)
#define MTOK   (16 * 96 * 120)    // 184320 tokens
#define NWIN   (MTOK / NTOK)      // 2304 windows

// ----- WMMA fragment loaders (CDNA5 16x16x32 f16 layouts, wave32) -----------

// A (16xK tile slice, row-major, lda in halfs): lane holds row m = lane&15.
// halfs 0..7  = K c0..c0+7 ; halfs 8..15 = K c0+16..c0+23 ; c0=(lane>>4)*8
__device__ __forceinline__ v16h frag_a(const _Float16* A, int lda, int lane) {
  const int m  = lane & 15;
  const int c0 = (lane >> 4) * 8;
  const _Float16* p = A + m * lda + c0;
  v8h lo = *(const v8h*)(p);
  v8h hi = *(const v8h*)(p + 16);
  v16h r;
#pragma unroll
  for (int i = 0; i < 8; ++i) { r[i] = lo[i]; r[i + 8] = hi[i]; }
  return r;
}

// B (Kx16 tile) supplied as W[n][k] row-major (ldw in halfs): lane holds
// column n = lane&15, halfs 0..7 = K c0..c0+7 ; halfs 8..15 = K c0+16..c0+23 ; c0=(lane>>4)*8
__device__ __forceinline__ v16h frag_b(const _Float16* W, int ldw, int lane) {
  const int n  = lane & 15;
  const int kb = (lane >> 4) * 8;
  const _Float16* p = W + n * ldw + kb;
  v8h lo = *(const v8h*)(p);
  v8h hi = *(const v8h*)(p + 16);
  v16h r;
#pragma unroll
  for (int i = 0; i < 8; ++i) { r[i] = lo[i]; r[i + 8] = hi[i]; }
  return r;
}

__device__ __forceinline__ v8f wmma16(v16h a, v16h b, v8f c) {
  return __builtin_amdgcn_wmma_f32_16x16x32_f16(
      false, a, false, b, (short)0, c, false, false);
}

__device__ __forceinline__ float gelu_exact(float x) {
  return 0.5f * x * (1.0f + erff(x * 0.70710678118654752f));
}

// Software-pipelined 32x64 wave tile: D[2][4] (two 16-row groups, four
// 16-col groups). Prefetches step k+32 fragments before the 8 WMMAs of step k.
template <int K>
__device__ __forceinline__ void wave_tile_32x64(
    const _Float16* __restrict__ Arow0, const _Float16* __restrict__ Arow1,
    int lda, const _Float16* __restrict__ Wbase, int ldw, int lane,
    v8f (&acc0)[4], v8f (&acc1)[4]) {
  v16h a0 = frag_a(Arow0, lda, lane);
  v16h a1 = frag_a(Arow1, lda, lane);
  v16h b[4];
#pragma unroll
  for (int t = 0; t < 4; ++t) b[t] = frag_b(Wbase + t * 16 * ldw, ldw, lane);
#pragma unroll
  for (int k = 0; k < K; k += 32) {
    v16h ca0 = a0, ca1 = a1;
    v16h cb0 = b[0], cb1 = b[1], cb2 = b[2], cb3 = b[3];
    if (k + 32 < K) {                       // prefetch next k-step
      a0 = frag_a(Arow0 + k + 32, lda, lane);
      a1 = frag_a(Arow1 + k + 32, lda, lane);
#pragma unroll
      for (int t = 0; t < 4; ++t)
        b[t] = frag_b(Wbase + t * 16 * ldw + k + 32, ldw, lane);
    }
    acc0[0] = wmma16(ca0, cb0, acc0[0]);
    acc1[0] = wmma16(ca1, cb0, acc1[0]);
    acc0[1] = wmma16(ca0, cb1, acc0[1]);
    acc1[1] = wmma16(ca1, cb1, acc1[1]);
    acc0[2] = wmma16(ca0, cb2, acc0[2]);
    acc1[2] = wmma16(ca1, cb2, acc1[2]);
    acc0[3] = wmma16(ca0, cb3, acc0[3]);
    acc1[3] = wmma16(ca1, cb3, acc1[3]);
  }
}

// ----- small helper kernels -------------------------------------------------

__global__ void k_f32_to_f16(const float* __restrict__ src,
                             _Float16* __restrict__ dst, int n) {
  int i = blockIdx.x * blockDim.x + threadIdx.x;
  if (i < n) dst[i] = (_Float16)src[i];
}

// CPB table: relu(table @ w1^T + b1) @ w2^T  -> tab[285][12]
__global__ void k_cpb_tab(const float* __restrict__ w1, const float* __restrict__ b1,
                          const float* __restrict__ w2, float* __restrict__ tab) {
  int t = blockIdx.x * blockDim.x + threadIdx.x;
  if (t >= 285) return;
  int i = t / 19, j = t % 19;
  float c0 = ((float)(i - (WHGT - 1)) / (float)(WHGT - 1)) * 8.0f;
  float c1 = ((float)(j - (WWID - 1)) / (float)(WWID - 1)) * 8.0f;
  const float il8 = 1.0f / log2f(8.0f);
  c0 = (c0 == 0.0f) ? 0.0f : copysignf(log2f(fabsf(c0) + 1.0f) * il8, c0);
  c1 = (c1 == 0.0f) ? 0.0f : copysignf(log2f(fabsf(c1) + 1.0f) * il8, c1);
  float acc[NHEADS];
#pragma unroll
  for (int h = 0; h < NHEADS; ++h) acc[h] = 0.0f;
  for (int k = 0; k < 512; ++k) {
    float hv = fmaxf(w1[k * 2 + 0] * c0 + w1[k * 2 + 1] * c1 + b1[k], 0.0f);
#pragma unroll
    for (int h = 0; h < NHEADS; ++h) acc[h] += w2[h * 512 + k] * hv;
  }
#pragma unroll
  for (int h = 0; h < NHEADS; ++h) tab[t * NHEADS + h] = acc[h];
}

// bias16[h][n][m] = 16 * sigmoid(tab[idx(n,m)][h])
__global__ void k_bias16(const float* __restrict__ tab, float* __restrict__ bias16) {
  int id = blockIdx.x * blockDim.x + threadIdx.x;
  if (id >= NHEADS * NTOK * NTOK) return;
  int h = id / (NTOK * NTOK);
  int n = (id / NTOK) % NTOK;
  int m = id % NTOK;
  int dr = n / WWID - m / WWID + (WHGT - 1);
  int dc = n % WWID - m % WWID + (WWID - 1);
  int idx = dr * (2 * WWID - 1) + dc;
  float v = tab[idx * NHEADS + h];
  bias16[id] = 16.0f / (1.0f + expf(-v));
}

// LayerNorm over C=384, output f16. windowed=1 -> window-partitioned row order.
__global__ __launch_bounds__(128)
void k_layernorm(const float* __restrict__ x, const float* __restrict__ g,
                 const float* __restrict__ b, _Float16* __restrict__ y, int windowed) {
  __shared__ float rs[128], rq[128];
  const int tok = blockIdx.x;
  const int tid = threadIdx.x;
  const float* xr = x + (size_t)tok * DIMC;
  float v[3], s = 0.0f, q = 0.0f;
#pragma unroll
  for (int i = 0; i < 3; ++i) {
    v[i] = xr[tid + i * 128];
    s += v[i];
    q += v[i] * v[i];
  }
  rs[tid] = s; rq[tid] = q;
  __syncthreads();
  for (int off = 64; off > 0; off >>= 1) {
    if (tid < off) { rs[tid] += rs[tid + off]; rq[tid] += rq[tid + off]; }
    __syncthreads();
  }
  const float mean = rs[0] * (1.0f / DIMC);
  const float var  = rq[0] * (1.0f / DIMC) - mean * mean;
  const float inv  = rsqrtf(var + 1e-5f);
  size_t row;
  if (windowed) {
    int bb = tok / (96 * 120);
    int sh = (tok / 120) % 96;
    int sw = tok % 120;
    int wr = sh / WHGT, r = sh % WHGT;
    int wc = sw / WWID, c = sw % WWID;
    row = ((size_t)((bb * 12 + wr) * 12 + wc)) * NTOK + (size_t)(r * WWID + c);
  } else {
    row = (size_t)tok;
  }
  _Float16* yr = y + row * DIMC;
#pragma unroll
  for (int i = 0; i < 3; ++i) {
    int c = tid + i * 128;
    yr[c] = (_Float16)((v[i] - mean) * inv * g[c] + b[c]);
  }
}

// ----- GEMM kernels: per-wave 32x64 tile, prefetch-pipelined ----------------

// QKV: out[M,1152] = Y[M,384] @ Wqkv^T + [q_bias, 0, v_bias]
__global__ __launch_bounds__(256)
void k_gemm_qkv(const _Float16* __restrict__ A, const _Float16* __restrict__ W,
                const float* __restrict__ qb, const float* __restrict__ vb,
                _Float16* __restrict__ out) {
  const int lane = threadIdx.x & 31;
  const int mtile = blockIdx.x * 8 + (threadIdx.x >> 5);   // 32 rows each
  const int nbase = blockIdx.y * 64;
  const _Float16* Arow0 = A + (size_t)mtile * 32 * DIMC;
  const _Float16* Arow1 = Arow0 + (size_t)16 * DIMC;
  v8f acc0[4] = {v8f{}, v8f{}, v8f{}, v8f{}};
  v8f acc1[4] = {v8f{}, v8f{}, v8f{}, v8f{}};
  wave_tile_32x64<DIMC>(Arow0, Arow1, DIMC, W + (size_t)nbase * DIMC, DIMC,
                        lane, acc0, acc1);
  const int hi = lane >> 4, nc = lane & 15;
#pragma unroll
  for (int t = 0; t < 4; ++t) {
    int n = nbase + t * 16 + nc;
    float bias = (n < DIMC) ? qb[n] : ((n < 2 * DIMC) ? 0.0f : vb[n - 2 * DIMC]);
#pragma unroll
    for (int r = 0; r < 8; ++r) {
      int m = mtile * 32 + r + 8 * hi;
      out[(size_t)m * (3 * DIMC) + n]        = (_Float16)(acc0[t][r] + bias);
      out[(size_t)(m + 16) * (3 * DIMC) + n] = (_Float16)(acc1[t][r] + bias);
    }
  }
}

// Proj: x1 = x + ls1 * (O[M,384] @ Wp^T + pb), un-windowing rows.
__global__ __launch_bounds__(256)
void k_gemm_proj(const _Float16* __restrict__ A, const _Float16* __restrict__ W,
                 const float* __restrict__ pb, const float* __restrict__ ls1,
                 const float* __restrict__ x, float* __restrict__ x1) {
  const int lane = threadIdx.x & 31;
  const int mtile = blockIdx.x * 8 + (threadIdx.x >> 5);
  const int nbase = blockIdx.y * 64;
  const _Float16* Arow0 = A + (size_t)mtile * 32 * DIMC;
  const _Float16* Arow1 = Arow0 + (size_t)16 * DIMC;
  v8f acc0[4] = {v8f{}, v8f{}, v8f{}, v8f{}};
  v8f acc1[4] = {v8f{}, v8f{}, v8f{}, v8f{}};
  wave_tile_32x64<DIMC>(Arow0, Arow1, DIMC, W + (size_t)nbase * DIMC, DIMC,
                        lane, acc0, acc1);
  const int hi = lane >> 4, nc = lane & 15;
#pragma unroll
  for (int t = 0; t < 4; ++t) {
    int n = nbase + t * 16 + nc;
    float bn = pb[n], sn = ls1[n];
#pragma unroll
    for (int g = 0; g < 2; ++g) {
#pragma unroll
      for (int r = 0; r < 8; ++r) {
        int m = mtile * 32 + g * 16 + r + 8 * hi;   // windowed row -> spatial
        int win = m / NTOK, tk = m % NTOK;
        int bb = win / 144, wr = (win / 12) % 12, wc = win % 12;
        int sh = wr * WHGT + tk / WWID;
        int sw = wc * WWID + tk % WWID;
        size_t gi = (((size_t)bb * 96 + sh) * 120 + sw) * DIMC + n;
        float av = g ? acc1[t][r] : acc0[t][r];
        x1[gi] = x[gi] + sn * (av + bn);
      }
    }
  }
}

// GLU: wave computes columns [na,na+64) and [na+1024,...): h = a * gelu(gate)
// 16 rows, 8 WMMA per k-step across the a/gate pair, prefetch-pipelined.
__global__ __launch_bounds__(256)
void k_gemm_glu(const _Float16* __restrict__ A, const _Float16* __restrict__ W,
                const float* __restrict__ gb, _Float16* __restrict__ h) {
  const int lane = threadIdx.x & 31;
  const int mtile = blockIdx.x * 8 + (threadIdx.x >> 5);
  const int na = blockIdx.y * 64;
  const int ng = na + 1024;
  const _Float16* Arow = A + (size_t)mtile * 16 * DIMC;
  const _Float16* WA = W + (size_t)na * DIMC;
  const _Float16* WG = W + (size_t)ng * DIMC;
  v8f accA[4] = {v8f{}, v8f{}, v8f{}, v8f{}};
  v8f accG[4] = {v8f{}, v8f{}, v8f{}, v8f{}};
  v16h a = frag_a(Arow, DIMC, lane);
  v16h bA[4], bG[4];
#pragma unroll
  for (int t = 0; t < 4; ++t) {
    bA[t] = frag_b(WA + t * 16 * DIMC, DIMC, lane);
    bG[t] = frag_b(WG + t * 16 * DIMC, DIMC, lane);
  }
#pragma unroll
  for (int k = 0; k < DIMC; k += 32) {
    v16h ca = a;
    v16h cA0 = bA[0], cA1 = bA[1], cA2 = bA[2], cA3 = bA[3];
    v16h cG0 = bG[0], cG1 = bG[1], cG2 = bG[2], cG3 = bG[3];
    if (k + 32 < DIMC) {
      a = frag_a(Arow + k + 32, DIMC, lane);
#pragma unroll
      for (int t = 0; t < 4; ++t) {
        bA[t] = frag_b(WA + t * 16 * DIMC + k + 32, DIMC, lane);
        bG[t] = frag_b(WG + t * 16 * DIMC + k + 32, DIMC, lane);
      }
    }
    accA[0] = wmma16(ca, cA0, accA[0]);
    accG[0] = wmma16(ca, cG0, accG[0]);
    accA[1] = wmma16(ca, cA1, accA[1]);
    accG[1] = wmma16(ca, cG1, accG[1]);
    accA[2] = wmma16(ca, cA2, accA[2]);
    accG[2] = wmma16(ca, cG2, accG[2]);
    accA[3] = wmma16(ca, cA3, accA[3]);
    accG[3] = wmma16(ca, cG3, accG[3]);
  }
  const int hi = lane >> 4, nc = lane & 15;
#pragma unroll
  for (int t = 0; t < 4; ++t) {
    int n = na + t * 16 + nc;
    float ba = gb[n], bg = gb[n + 1024];
#pragma unroll
    for (int r = 0; r < 8; ++r) {
      int m = mtile * 16 + r + 8 * hi;
      float av = accA[t][r] + ba;
      float gv = accG[t][r] + bg;
      h[(size_t)m * 1024 + n] = (_Float16)(av * gelu_exact(gv));
    }
  }
}

// Out: out = x1 + ls2 * (H[M,1024] @ Wo^T + ob)
__global__ __launch_bounds__(256)
void k_gemm_out(const _Float16* __restrict__ A, const _Float16* __restrict__ W,
                const float* __restrict__ ob, const float* __restrict__ ls2,
                const float* __restrict__ x1, float* __restrict__ out) {
  const int lane = threadIdx.x & 31;
  const int mtile = blockIdx.x * 8 + (threadIdx.x >> 5);
  const int nbase = blockIdx.y * 64;
  const _Float16* Arow0 = A + (size_t)mtile * 32 * 1024;
  const _Float16* Arow1 = Arow0 + (size_t)16 * 1024;
  v8f acc0[4] = {v8f{}, v8f{}, v8f{}, v8f{}};
  v8f acc1[4] = {v8f{}, v8f{}, v8f{}, v8f{}};
  wave_tile_32x64<1024>(Arow0, Arow1, 1024, W + (size_t)nbase * 1024, 1024,
                        lane, acc0, acc1);
  const int hi = lane >> 4, nc = lane & 15;
#pragma unroll
  for (int t = 0; t < 4; ++t) {
    int n = nbase + t * 16 + nc;
    float bn = ob[n], sn = ls2[n];
#pragma unroll
    for (int r = 0; r < 8; ++r) {
      int m = mtile * 32 + r + 8 * hi;
      size_t gi0 = (size_t)m * DIMC + n;
      size_t gi1 = (size_t)(m + 16) * DIMC + n;
      out[gi0] = x1[gi0] + sn * (acc0[t][r] + bn);
      out[gi1] = x1[gi1] + sn * (acc1[t][r] + bn);
    }
  }
}

// ----- fused attention: one block per (window, head) ------------------------
// cosine attention (row-normalized q,k), bias = 16*sigmoid(cpb), softmax, AV.
__global__ __launch_bounds__(256)
void k_attention(const _Float16* __restrict__ qkv, const float* __restrict__ bias16,
                 _Float16* __restrict__ o) {
  __shared__ __align__(32) _Float16 qs[NTOK * HDIM];       // 5120 B
  __shared__ __align__(32) _Float16 ks[NTOK * HDIM];       // 5120 B
  __shared__ __align__(32) _Float16 vt[HDIM * 96];         // 6144 B (K padded to 96)
  __shared__ __align__(32) float    sc[NTOK * NTOK];       // 25600 B
  __shared__ __align__(32) _Float16 ah[NTOK * 96];         // 15360 B
  const int win = blockIdx.x;
  const int head = blockIdx.y;
  const int tid = threadIdx.x;
  const _Float16* base = qkv + (size_t)win * NTOK * (3 * DIMC) + head * HDIM;

  // normalize q,k rows (5 waves), load v transposed with zero K-padding
  if (tid < 2 * NTOK) {
    int which = tid / NTOK;                 // 0 = q, 1 = k
    int t = tid % NTOK;
    const _Float16* p = base + (size_t)t * (3 * DIMC) + which * DIMC;
    float tmp[HDIM], ss = 0.0f;
#pragma unroll
    for (int d = 0; d < HDIM; ++d) { tmp[d] = (float)p[d]; ss += tmp[d] * tmp[d]; }
    float s = 1.0f / (sqrtf(ss) + 5e-5f);
    _Float16* dst = (which ? ks : qs) + t * HDIM;
#pragma unroll
    for (int d = 0; d < HDIM; ++d) dst[d] = (_Float16)(tmp[d] * s);
  }
  for (int i = tid; i < HDIM * 96; i += 256) {
    int d = i / 96, t = i % 96;
    float val = (t < NTOK) ? (float)base[(size_t)t * (3 * DIMC) + 2 * DIMC + d] : 0.0f;
    vt[d * 96 + t] = (_Float16)val;
  }
  __syncthreads();

  const int wv = tid >> 5, lane = tid & 31;

  // scores: 5x5 tiles of 16x16, K = 32 (one WMMA each)
  for (int tt = wv; tt < 25; tt += 8) {
    int ti = tt / 5, tj = tt % 5;
    v16h a = frag_a(qs + ti * 16 * HDIM, HDIM, lane);
    v16h b = frag_b(ks + tj * 16 * HDIM, HDIM, lane);
    v8f c = wmma16(a, b, v8f{});
    const int hi = lane >> 4, nc = lane & 15;
#pragma unroll
    for (int r = 0; r < 8; ++r)
      sc[(ti * 16 + r + 8 * hi) * NTOK + tj * 16 + nc] = c[r];
  }
  __syncthreads();

  // softmax(scores + 16*sigmoid(bias)) per query row
  if (tid < NTOK) {
    const float* bb = bias16 + (size_t)head * NTOK * NTOK + tid * NTOK;
    float* row = sc + tid * NTOK;
    float mx = -1e30f;
    for (int j = 0; j < NTOK; ++j) {
      float v = row[j] + bb[j];
      row[j] = v;
      mx = fmaxf(mx, v);
    }
    float sum = 0.0f;
    for (int j = 0; j < NTOK; ++j) { float e = expf(row[j] - mx); row[j] = e; sum += e; }
    float inv = 1.0f / sum;
    for (int j = 0; j < NTOK; ++j) ah[tid * 96 + j] = (_Float16)(row[j] * inv);
#pragma unroll
    for (int j = NTOK; j < 96; ++j) ah[tid * 96 + j] = (_Float16)0.0f;
  }
  __syncthreads();

  // O = attn @ V : 5 m-tiles x 2 n-tiles, K padded to 96 (3 WMMA each)
  for (int tt = wv; tt < 10; tt += 8) {
    int ti = tt / 2, tj = tt % 2;
    v8f c = v8f{};
    for (int k = 0; k < 96; k += 32) {
      v16h a = frag_a(ah + ti * 16 * 96 + k, 96, lane);
      v16h b = frag_b(vt + tj * 16 * 96 + k, 96, lane);
      c = wmma16(a, b, c);
    }
    const int hi = lane >> 4, nc = lane & 15;
#pragma unroll
    for (int r = 0; r < 8; ++r) {
      int m = ti * 16 + r + 8 * hi;
      int d = tj * 16 + nc;
      o[((size_t)win * NTOK + m) * DIMC + head * HDIM + d] = (_Float16)c[r];
    }
  }
}

// ---------------------------------------------------------------------------

extern "C" void kernel_launch(void* const* d_in, const int* in_sizes, int n_in,
                              void* d_out, int out_size, void* d_ws, size_t ws_size,
                              hipStream_t stream) {
  (void)in_sizes; (void)n_in; (void)out_size; (void)ws_size;
  const float* x      = (const float*)d_in[0];
  const float* n1g    = (const float*)d_in[1];
  const float* n1b    = (const float*)d_in[2];
  const float* qkv_w  = (const float*)d_in[3];
  const float* q_bias = (const float*)d_in[4];
  const float* v_bias = (const float*)d_in[5];
  const float* cpb_w1 = (const float*)d_in[6];
  const float* cpb_b1 = (const float*)d_in[7];
  const float* cpb_w2 = (const float*)d_in[8];
  const float* proj_w = (const float*)d_in[9];
  const float* proj_b = (const float*)d_in[10];
  const float* ls1_g  = (const float*)d_in[11];
  const float* n2g    = (const float*)d_in[12];
  const float* n2b    = (const float*)d_in[13];
  const float* glu_w  = (const float*)d_in[14];
  const float* glu_b  = (const float*)d_in[15];
  const float* out_w  = (const float*)d_in[16];
  const float* out_b  = (const float*)d_in[17];
  const float* ls2_g  = (const float*)d_in[18];
  float* out = (float*)d_out;

  // workspace carve-up (256B aligned)
  char* ws = (char*)d_ws;
  size_t off = 0;
  auto carve = [&](size_t bytes) -> char* {
    char* p = ws + off;
    off = (off + bytes + 255) & ~(size_t)255;
    return p;
  };
  _Float16* wq   = (_Float16*)carve((size_t)3 * DIMC * DIMC * 2);       // 1152x384
  _Float16* wp   = (_Float16*)carve((size_t)DIMC * DIMC * 2);           // 384x384
  _Float16* wg   = (_Float16*)carve((size_t)2048 * DIMC * 2);           // 2048x384
  _Float16* wo   = (_Float16*)carve((size_t)DIMC * 1024 * 2);           // 384x1024
  float*    tab  = (float*)carve((size_t)285 * NHEADS * 4);
  float*    b16  = (float*)carve((size_t)NHEADS * NTOK * NTOK * 4);
  _Float16* ybuf = (_Float16*)carve((size_t)MTOK * DIMC * 2);           // LN out (reused)
  _Float16* qkvb = (_Float16*)carve((size_t)MTOK * 3 * DIMC * 2);       // qkv (reused as h)
  _Float16* obuf = (_Float16*)carve((size_t)MTOK * DIMC * 2);           // attn out
  float*    x1   = (float*)carve((size_t)MTOK * DIMC * 4);              // residual 1
  _Float16* hbuf = qkvb;   // GLU hidden reuses dead qkv region (378MB <= 425MB)

  // 1) weight conversions to f16 (L2-resident afterwards)
  {
    int n;
    n = 3 * DIMC * DIMC; k_f32_to_f16<<<(n + 255) / 256, 256, 0, stream>>>(qkv_w, wq, n);
    n = DIMC * DIMC;     k_f32_to_f16<<<(n + 255) / 256, 256, 0, stream>>>(proj_w, wp, n);
    n = 2048 * DIMC;     k_f32_to_f16<<<(n + 255) / 256, 256, 0, stream>>>(glu_w, wg, n);
    n = DIMC * 1024;     k_f32_to_f16<<<(n + 255) / 256, 256, 0, stream>>>(out_w, wo, n);
  }
  // 2) continuous relative position bias (tiny)
  k_cpb_tab<<<2, 256, 0, stream>>>(cpb_w1, cpb_b1, cpb_w2, tab);
  k_bias16<<<(NHEADS * NTOK * NTOK + 255) / 256, 256, 0, stream>>>(tab, b16);
  // 3) LN1 -> windowed f16
  k_layernorm<<<MTOK, 128, 0, stream>>>(x, n1g, n1b, ybuf, 1);
  // 4) QKV GEMM (M x 384 -> M x 1152), 32-row wave tiles
  k_gemm_qkv<<<dim3(MTOK / 256, (3 * DIMC) / 64), 256, 0, stream>>>(
      ybuf, wq, q_bias, v_bias, qkvb);
  // 5) fused window attention
  k_attention<<<dim3(NWIN, NHEADS), 256, 0, stream>>>(qkvb, b16, obuf);
  // 6) proj GEMM + un-window + residual*ls1 -> x1
  k_gemm_proj<<<dim3(MTOK / 256, DIMC / 64), 256, 0, stream>>>(
      obuf, wp, proj_b, ls1_g, x, x1);
  // 7) LN2 -> token-order f16
  k_layernorm<<<MTOK, 128, 0, stream>>>(x1, n2g, n2b, ybuf, 0);
  // 8) GLU GEMM (paired a/gate tiles, fused gelu gating) -> h (M x 1024)
  k_gemm_glu<<<dim3(MTOK / 128, 1024 / 64), 256, 0, stream>>>(ybuf, wg, glu_b, hbuf);
  // 9) out GEMM + residual*ls2 -> d_out
  k_gemm_out<<<dim3(MTOK / 256, DIMC / 64), 256, 0, stream>>>(
      hbuf, wo, out_b, ls2_g, x1, out);
}
